// _MultiHeadGraphAttentionLayer_47210280518130
// MI455X (gfx1250) — matrix operand order repaired and checked by reading, not compiled, not run on hardware
//
#include <hip/hip_runtime.h>
#include <hip/hip_bf16.h>

typedef __attribute__((ext_vector_type(16))) __bf16 v16bf;
typedef __attribute__((ext_vector_type(8)))  float  v8f;

#define N_NODES 4096
#define FIN     256
#define FOUT    128
#define NHEAD   4
#define KCAT    (NHEAD * FOUT)   // 512
#define NEG_BIG (-9.0e15f)

// ---------------------------------------------------------------------------
// Kernel 0: pack W -> WT[h][o][i] (bf16) and W_out -> WoT[o][k] (bf16)
// so WMMA B-operands become contiguous 32-byte loads.
// ---------------------------------------------------------------------------
__global__ void gat_pack_weights(const float* __restrict__ W,
                                 const float* __restrict__ Wout,
                                 __bf16* __restrict__ WT,
                                 __bf16* __restrict__ WoT) {
    int idx = blockIdx.x * blockDim.x + threadIdx.x;
    const int nW = NHEAD * FIN * FOUT;
    if (idx < nW) {
        int hh  = idx / (FIN * FOUT);
        int rem = idx - hh * (FIN * FOUT);
        int i   = rem / FOUT;
        int o   = rem - i * FOUT;
        WT[((size_t)hh * FOUT + o) * FIN + i] = (__bf16)W[idx];
    } else {
        int j = idx - nW;
        if (j < KCAT * FOUT) {
            int k = j / FOUT;
            int o = j - k * FOUT;
            WoT[(size_t)o * KCAT + k] = (__bf16)Wout[j];
        }
    }
}

// ---------------------------------------------------------------------------
// Kernel 1: Wh = h @ W[head]  (4096x256)@(256x128) per head, bf16 WMMA.
// One wave per (16-row tile, head). Stores Wh transposed: WhbT[h][o][n] bf16.
// B tiles are staged into a register array first so the backend can clause
// the 16 b128 loads and overlap them with the WMMA burst.
// ---------------------------------------------------------------------------
__global__ __launch_bounds__(32)
void gat_wh_gemm(const float* __restrict__ h,
                 const __bf16* __restrict__ WT,
                 __bf16* __restrict__ WhbT) {
    const int hh    = blockIdx.y;
    const int rbase = blockIdx.x * 16;
    const int lane  = threadIdx.x;
    const int r     = lane & 15;
    const int half  = lane >> 4;
    const int kb    = half * 8;

    const v8f vzero = {0.f, 0.f, 0.f, 0.f, 0.f, 0.f, 0.f, 0.f};
    v8f acc[8];
#pragma unroll
    for (int t = 0; t < 8; ++t) acc[t] = vzero;

    const float*  hrow   = h + (size_t)(rbase + r) * FIN;
    const __bf16* wthead = WT + (size_t)hh * FOUT * FIN;

    for (int k0 = 0; k0 < FIN; k0 += 32) {
        // stage all B tiles (issued as one load clause)
        v16bf Bt[8];
#pragma unroll
        for (int t = 0; t < 8; ++t)
            Bt[t] = *(const v16bf*)(wthead + (size_t)(t * 16 + r) * FIN + k0 + half * 16);

        float4 a0 = *(const float4*)(hrow + k0 + half * 16);
        float4 a1 = *(const float4*)(hrow + k0 + half * 16 + 4);
        float4 a2 = *(const float4*)(hrow + k0 + half * 16 + 8);
        float4 a3 = *(const float4*)(hrow + k0 + half * 16 + 12);
        float av[16] = {a0.x, a0.y, a0.z, a0.w, a1.x, a1.y, a1.z, a1.w,
                        a2.x, a2.y, a2.z, a2.w, a3.x, a3.y, a3.z, a3.w};
        v16bf A;
#pragma unroll
        for (int i = 0; i < 16; ++i) A[i] = (__bf16)av[i];

#pragma unroll
        for (int t = 0; t < 8; ++t)
            acc[t] = __builtin_amdgcn_wmma_f32_16x16x32_bf16(
                false, A, false, Bt[t], (short)0, acc[t], false, false);
    }

#pragma unroll
    for (int t = 0; t < 8; ++t) {
#pragma unroll
        for (int j = 0; j < 8; ++j) {
            const int col  = t * 16 + r;
            const int rowg = rbase + j + half * 8;
            WhbT[((size_t)hh * FOUT + col) * N_NODES + rowg] = (__bf16)acc[t][j];
        }
    }
}

// ---------------------------------------------------------------------------
// Kernel 2: f[h][n] = Wh[h][n]·a1[h],  g[h][n] = Wh[h][n]·a2[h]
// ---------------------------------------------------------------------------
__global__ void gat_fg(const __bf16* __restrict__ WhbT,
                       const float* __restrict__ a,
                       float* __restrict__ f,
                       float* __restrict__ g) {
    const int idx = blockIdx.x * blockDim.x + threadIdx.x;   // [0, H*N)
    const int n   = idx & (N_NODES - 1);
    const int hh  = idx >> 12;
    const __bf16* base = WhbT + (size_t)hh * FOUT * N_NODES + n;
    const float*  a1   = a + hh * 2 * FOUT;
    const float*  a2   = a1 + FOUT;
    float fs = 0.f, gs = 0.f;
#pragma unroll 4
    for (int o = 0; o < FOUT; ++o) {
        float wh = (float)base[(size_t)o * N_NODES];
        fs += wh * a1[o];
        gs += wh * a2[o];
    }
    f[idx] = fs;
    g[idx] = gs;
}

// ---------------------------------------------------------------------------
// Kernel 3: fused masked attention + softmax + PV (flash-style, online).
// Block = 128 threads = 4 waves; wave w handles head w for the same 16-row
// tile, so each adj tile is reused 4x from WGP$/L2. PV via bf16 WMMA.
// B tiles are loaded right after adj/g so the 16 b128 loads are in flight
// while the softmax VALU/transcendental work executes.
// Writes hp in "cat" layout: hp[n][h*FOUT + o] fp32, with ELU applied.
// ---------------------------------------------------------------------------
__global__ __launch_bounds__(128)
void gat_attn(const int* __restrict__ adj,
              const __bf16* __restrict__ WhbT,
              const float* __restrict__ f,
              const float* __restrict__ g,
              float* __restrict__ hp) {
    const int rbase = blockIdx.x * 16;
    const int hh    = threadIdx.x >> 5;
    const int lane  = threadIdx.x & 31;
    const int r     = lane & 15;
    const int half  = lane >> 4;
    const int kb    = half * 8;
    const int rowg  = rbase + r;

    const float   fval = f[hh * N_NODES + rowg];
    const float*  grow = g + hh * N_NODES;
    const int*    arow = adj + (size_t)rowg * N_NODES;
    const __bf16* whb  = WhbT + (size_t)hh * FOUT * N_NODES;

    float mrun = -INFINITY;
    float lrun = 0.0f;
    const v8f vzero = {0.f, 0.f, 0.f, 0.f, 0.f, 0.f, 0.f, 0.f};
    v8f acc[8];
#pragma unroll
    for (int t = 0; t < 8; ++t) acc[t] = vzero;

    for (int mb = 0; mb < N_NODES; mb += 32) {
        // ---- issue all loads for this chunk up front ----
        int4   ad0 = *(const int4*)(arow + mb + half * 16);
        int4   ad1 = *(const int4*)(arow + mb + half * 16 + 4);
        int4   ad2 = *(const int4*)(arow + mb + half * 16 + 8);
        int4   ad3 = *(const int4*)(arow + mb + half * 16 + 12);
        float4 g0  = *(const float4*)(grow + mb + half * 16);
        float4 g1  = *(const float4*)(grow + mb + half * 16 + 4);
        float4 g2  = *(const float4*)(grow + mb + half * 16 + 8);
        float4 g3  = *(const float4*)(grow + mb + half * 16 + 12);

        v16bf Bt[8];
#pragma unroll
        for (int t = 0; t < 8; ++t)
            Bt[t] = *(const v16bf*)(whb + (size_t)(t * 16 + r) * N_NODES + mb + half * 16);

        if (mb + 32 < N_NODES)
            __builtin_prefetch(arow + mb + 32 + kb, 0, 1);   // global_prefetch_b8

        // ---- softmax VALU work overlaps the in-flight B-tile loads ----
        int   adv[16] = {ad0.x, ad0.y, ad0.z, ad0.w, ad1.x, ad1.y, ad1.z, ad1.w,
                         ad2.x, ad2.y, ad2.z, ad2.w, ad3.x, ad3.y, ad3.z, ad3.w};
        float gv[16]  = {g0.x, g0.y, g0.z, g0.w, g1.x, g1.y, g1.z, g1.w,
                         g2.x, g2.y, g2.z, g2.w, g3.x, g3.y, g3.z, g3.w};
        float ev[16];
#pragma unroll
        for (int i = 0; i < 16; ++i) {
            float x = fval + gv[i];
            x = (x > 0.f) ? x : 0.2f * x;            // leaky_relu(0.2)
            ev[i] = (adv[i] > 0) ? x : NEG_BIG;      // mask
        }

        // chunk max over this row's 32 columns (split across lane pair L,L+16)
        float cm = ev[0];
#pragma unroll
        for (int i = 1; i < 16; ++i) cm = fmaxf(cm, ev[i]);
        cm = fmaxf(cm, __shfl_xor(cm, 16, 32));
        const float mnew = fmaxf(mrun, cm);
        const float sc   = __expf(mrun - mnew);      // exp(-inf)=0 on first chunk

        float psum = 0.f;
        v16bf pA;
#pragma unroll
        for (int i = 0; i < 16; ++i) {
            float p = __expf(ev[i] - mnew);
            psum += p;
            pA[i] = (__bf16)p;
        }
        psum += __shfl_xor(psum, 16, 32);
        lrun = lrun * sc + psum;
        mrun = mnew;

        // broadcast per-row rescale factor into the C/D register layout
        float scj[8];
#pragma unroll
        for (int j = 0; j < 8; ++j) scj[j] = __shfl(sc, j + half * 8, 32);
#pragma unroll
        for (int t = 0; t < 8; ++t) {
#pragma unroll
            for (int j = 0; j < 8; ++j) acc[t][j] *= scj[j];
        }

        // ---- PV: acc += P(16x32,bf16) @ Wh(32x16,bf16) per column tile ----
#pragma unroll
        for (int t = 0; t < 8; ++t)
            acc[t] = __builtin_amdgcn_wmma_f32_16x16x32_bf16(
                false, pA, false, Bt[t], (short)0, acc[t], false, false);
    }

    const float linv = 1.0f / lrun;
    float lj[8];
#pragma unroll
    for (int j = 0; j < 8; ++j) lj[j] = __shfl(linv, j + half * 8, 32);

#pragma unroll
    for (int t = 0; t < 8; ++t) {
#pragma unroll
        for (int j = 0; j < 8; ++j) {
            float v = acc[t][j] * lj[j];
            v = (v > 0.f) ? v : (__expf(v) - 1.0f);  // ELU
            const int nrow = rbase + j + half * 8;
            hp[(size_t)nrow * KCAT + hh * FOUT + t * 16 + r] = v;
        }
    }
}

// ---------------------------------------------------------------------------
// Kernel 4: out = elu(hp_cat @ W_out)  (4096x512)@(512x128), bf16 WMMA.
// ---------------------------------------------------------------------------
__global__ __launch_bounds__(32)
void gat_out_gemm(const float* __restrict__ hp,
                  const __bf16* __restrict__ WoT,
                  float* __restrict__ out) {
    const int rbase = blockIdx.x * 16;
    const int lane  = threadIdx.x;
    const int r     = lane & 15;
    const int half  = lane >> 4;
    const int kb    = half * 8;

    const v8f vzero = {0.f, 0.f, 0.f, 0.f, 0.f, 0.f, 0.f, 0.f};
    v8f acc[8];
#pragma unroll
    for (int t = 0; t < 8; ++t) acc[t] = vzero;

    const float* hrow = hp + (size_t)(rbase + r) * KCAT;

    for (int k0 = 0; k0 < KCAT; k0 += 32) {
        v16bf Bt[8];
#pragma unroll
        for (int t = 0; t < 8; ++t)
            Bt[t] = *(const v16bf*)(WoT + (size_t)(t * 16 + r) * KCAT + k0 + half * 16);

        float4 a0 = *(const float4*)(hrow + k0 + half * 16);
        float4 a1 = *(const float4*)(hrow + k0 + half * 16 + 4);
        float4 a2 = *(const float4*)(hrow + k0 + half * 16 + 8);
        float4 a3 = *(const float4*)(hrow + k0 + half * 16 + 12);
        float av[16] = {a0.x, a0.y, a0.z, a0.w, a1.x, a1.y, a1.z, a1.w,
                        a2.x, a2.y, a2.z, a2.w, a3.x, a3.y, a3.z, a3.w};
        v16bf A;
#pragma unroll
        for (int i = 0; i < 16; ++i) A[i] = (__bf16)av[i];

#pragma unroll
        for (int t = 0; t < 8; ++t)
            acc[t] = __builtin_amdgcn_wmma_f32_16x16x32_bf16(
                false, A, false, Bt[t], (short)0, acc[t], false, false);
    }

#pragma unroll
    for (int t = 0; t < 8; ++t) {
#pragma unroll
        for (int j = 0; j < 8; ++j) {
            float v = acc[t][j];
            v = (v > 0.f) ? v : (__expf(v) - 1.0f);  // ELU
            const int nrow = rbase + j + half * 8;
            out[(size_t)nrow * FOUT + t * 16 + r] = v;
        }
    }
}

// ---------------------------------------------------------------------------
// Launch
// ---------------------------------------------------------------------------
extern "C" void kernel_launch(void* const* d_in, const int* in_sizes, int n_in,
                              void* d_out, int out_size, void* d_ws, size_t ws_size,
                              hipStream_t stream) {
    (void)in_sizes; (void)n_in; (void)out_size; (void)ws_size;

    const float* h    = (const float*)d_in[0];
    const int*   adj  = (const int*)d_in[1];
    const float* W    = (const float*)d_in[2];
    const float* a    = (const float*)d_in[3];
    const float* Wout = (const float*)d_in[4];
    float*       out  = (float*)d_out;

    char*  ws  = (char*)d_ws;
    size_t off = 0;
    auto alloc = [&](size_t bytes) -> void* {
        void* p = ws + off;
        off += (bytes + 255) & ~(size_t)255;
        return p;
    };

    __bf16* WT    = (__bf16*)alloc((size_t)NHEAD * FIN * FOUT * 2);     // 256 KB
    __bf16* WoT   = (__bf16*)alloc((size_t)KCAT * FOUT * 2);            // 128 KB
    __bf16* WhbT  = (__bf16*)alloc((size_t)NHEAD * FOUT * N_NODES * 2); // 4 MB
    float*  fbuf  = (float*)alloc((size_t)NHEAD * N_NODES * 4);         // 64 KB
    float*  gbuf  = (float*)alloc((size_t)NHEAD * N_NODES * 4);         // 64 KB
    float*  hp    = (float*)alloc((size_t)N_NODES * KCAT * 4);          // 8 MB

    const int packN = NHEAD * FIN * FOUT + KCAT * FOUT;
    gat_pack_weights<<<(packN + 255) / 256, 256, 0, stream>>>(W, Wout, WT, WoT);
    gat_wh_gemm<<<dim3(N_NODES / 16, NHEAD), 32, 0, stream>>>(h, WT, WhbT);
    gat_fg<<<(NHEAD * N_NODES) / 256, 256, 0, stream>>>(WhbT, a, fbuf, gbuf);
    gat_attn<<<N_NODES / 16, 128, 0, stream>>>(adj, WhbT, fbuf, gbuf, hp);
    gat_out_gemm<<<N_NODES / 16, 32, 0, stream>>>(hp, WoT, out);
}
